// MultiHeadAttention_84628035600685
// MI455X (gfx1250) — matrix operand order repaired and checked by reading, not compiled, not run on hardware
//
#include <hip/hip_runtime.h>

// Problem constants (from reference)
#define BB   8
#define SS   1024
#define DD   1024
#define HH   16
#define DKV  64
// TEMPER = sqrt(D) = 32

typedef __bf16 bf16;
typedef __attribute__((ext_vector_type(16))) __bf16 bf16x16;
typedef __attribute__((ext_vector_type(8)))  float  f32x8;

// ---------------------------------------------------------------------------
// WMMA helpers (CDNA5 16x16x32 bf16, f32 accumulate).
// Fragment layouts per CDNA5 ISA 7.12.2 (wave32):
//  A (16x32 bf16): lane m=l&15; lanes 0-15 hold K {0..7,16..23}, lanes 16-31
//                  hold K {8..15,24..31}; each VGPR = 2 consecutive K.
//  B (32x16 bf16): lane n=l&15; K split as in A (lanes 0-15 K {0..7,16..23}, 16-31 K {8..15,24..31}).
//  C/D (16x16 f32): lane n=l&15; VGPR r holds row m = r + 8*(l>=16).
// ---------------------------------------------------------------------------
static __device__ __forceinline__ f32x8 wmma_bf16(bf16x16 a, bf16x16 b, f32x8 c) {
  return __builtin_amdgcn_wmma_f32_16x16x32_bf16(false, a, false, b, (short)0, c,
                                                 false, false);
}

// A fragment from row-major X[16][...], leading dim ld (elements), K offset kk.
static __device__ __forceinline__ bf16x16 load_a_frag(const bf16* base, int ld,
                                                      int kk) {
  int lane = threadIdx.x & 31;
  int m    = lane & 15;
  int kb2  = (lane >> 4) << 2;  // dword offset: 0 or 4 (K offset 0 or 8)
  union { bf16x16 v; unsigned u[8]; } f;
  const unsigned* p = (const unsigned*)(base + (size_t)m * ld + kk);
#pragma unroll
  for (int i = 0; i < 8; ++i)
    f.u[i] = p[kb2 + ((i >> 2) << 3) + (i & 3)];
  return f.v;
}

// B fragment with B[k][n] = X[n][kk+k] (k order as in A); X row-major, leading dim ld.
static __device__ __forceinline__ bf16x16 load_bt_frag(const bf16* base, int ld,
                                                       int kk) {
  int lane = threadIdx.x & 31;
  int n    = lane & 15;
  int koff = (lane >> 4) << 3;  // 0 or 8
  union { bf16x16 v; unsigned u[8]; } f;
  const unsigned* p = (const unsigned*)(base + (size_t)n * ld + kk + koff);
#pragma unroll
  for (int i = 0; i < 8; ++i) f.u[i] = p[((i >> 2) << 3) + (i & 3)];
  return f.v;
}

// ---------------------------------------------------------------------------
// 1) f32 -> bf16 converts
// ---------------------------------------------------------------------------
__global__ void __launch_bounds__(256)
cvt_f32_bf16(const float* __restrict__ in, bf16* __restrict__ out, int n) {
  int stride = gridDim.x * blockDim.x;
  for (int i = blockIdx.x * blockDim.x + threadIdx.x; i < n; i += stride)
    out[i] = (bf16)in[i];
}

// w[h][d][n] (H,D,64) -> wt[h][n][d] (H,64,D), bf16
__global__ void __launch_bounds__(256)
cvt_w_transpose(const float* __restrict__ in, bf16* __restrict__ out) {
  const int total = HH * DD * DKV;
  int stride = gridDim.x * blockDim.x;
  for (int i = blockIdx.x * blockDim.x + threadIdx.x; i < total; i += stride) {
    int h = i / (DD * DKV);
    int r = i % (DD * DKV);
    int d = r / DKV;
    int n = r % DKV;
    out[((size_t)h * DKV + n) * DD + d] = (bf16)in[i];
  }
}

// ---------------------------------------------------------------------------
// 2) Per-head input projection: Y[h,b,s,:] = X[b,s,:] @ W[h]   (K=1024, N=64)
//    Body = load A + all 4 B fragments (fresh SSA values), then 4 WMMAs.
//    In-order load returns give staggered partial waits (wmma j runs while
//    fragments j+1.. are still in flight) without loop-carried registers.
// ---------------------------------------------------------------------------
__global__ void __launch_bounds__(128)
head_proj_gemm(const bf16* __restrict__ X, const bf16* __restrict__ Wt,
               bf16* __restrict__ Y, int transposed) {
  int wave = threadIdx.x >> 5;
  int r0   = (blockIdx.x * (blockDim.x >> 5) + wave) * 16;  // row tile in B*S
  int h    = blockIdx.y;
  const bf16* Xt = X + (size_t)r0 * DD;
  const bf16* Wh = Wt + (size_t)h * DKV * DD;

  f32x8 zero = {};
  f32x8 acc[4];
#pragma unroll
  for (int j = 0; j < 4; ++j) acc[j] = zero;

#pragma unroll 2
  for (int kk = 0; kk < DD; kk += 32) {
    bf16x16 a = load_a_frag(Xt, DD, kk);
    bf16x16 b[4];
#pragma unroll
    for (int j = 0; j < 4; ++j)
      b[j] = load_bt_frag(Wh + (size_t)(j * 16) * DD, DD, kk);
#pragma unroll
    for (int j = 0; j < 4; ++j)
      acc[j] = wmma_bf16(a, b[j], acc[j]);
  }

  int lane = threadIdx.x & 31;
  int n = lane & 15, m0 = (lane >> 4) << 3;
  int bidx = r0 / SS, s0 = r0 % SS;
  if (!transposed) {
    bf16* base = Y + ((size_t)(h * BB + bidx) * SS + s0) * DKV;
#pragma unroll
    for (int j = 0; j < 4; ++j)
#pragma unroll
      for (int r = 0; r < 8; ++r)
        base[(size_t)(m0 + r) * DKV + j * 16 + n] = (bf16)acc[j][r];
  } else {
    bf16* base = Y + (size_t)(h * BB + bidx) * DKV * SS;
#pragma unroll
    for (int j = 0; j < 4; ++j)
#pragma unroll
      for (int r = 0; r < 8; ++r)
        base[(size_t)(j * 16 + n) * SS + s0 + m0 + r] = (bf16)acc[j][r];
  }
}

// ---------------------------------------------------------------------------
// 3) Attention: per (h,b,q-tile of 16): scores -> mask -> softmax -> P@V
//    Two waves per workgroup share one q-tile (~99KB LDS -> 6 waves/WGP):
//    waves split score tiles (even/odd), softmax row-quarters, and P@V
//    N-tiles (2 each). Cross-wave reductions via a small LDS array.
// ---------------------------------------------------------------------------
#define SSTR 1028  // f32 score row stride (pad: halves hit disjoint banks)
#define PSTR 1032  // bf16 prob row stride

__global__ void __launch_bounds__(64)
attn_kernel(const bf16* __restrict__ q_s, const bf16* __restrict__ k_s,
            const bf16* __restrict__ v_t, const int* __restrict__ mask,
            float* __restrict__ attns, bf16* __restrict__ out_heads) {
  __shared__ float sc[16 * SSTR];
  __shared__ bf16  pr[16 * PSTR];
  __shared__ float redmx[2][16];
  __shared__ float redsm[2][16];

  int q0 = blockIdx.x * 16;
  int b  = blockIdx.y;
  int h  = blockIdx.z;
  int tid  = threadIdx.x;
  int wave = tid >> 5;
  int lane = tid & 31;
  size_t hb = (size_t)(h * BB + b);

  const bf16* qs = q_s + (hb * SS + q0) * DKV;   // [16][64] tile
  const bf16* ks = k_s + hb * SS * DKV;          // [S][64]
  const bf16* vt = v_t + hb * DKV * SS;          // [64][S]

  // ---- scores: Q (16x64) @ K^T; waves interleave the 64 key tiles ----
  bf16x16 a0 = load_a_frag(qs, DKV, 0);
  bf16x16 a1 = load_a_frag(qs, DKV, 32);
  int n = lane & 15, m0 = (lane >> 4) << 3;
#pragma unroll 4
  for (int t = wave; t < SS / 16; t += 2) {
    const bf16* kt = ks + (size_t)(t * 16) * DKV;
    bf16x16 f0 = load_bt_frag(kt, DKV, 0);
    bf16x16 f1 = load_bt_frag(kt, DKV, 32);
    f32x8 acc = {};
    acc = wmma_bf16(a0, f0, acc);
    acc = wmma_bf16(a1, f1, acc);
#pragma unroll
    for (int r = 0; r < 8; ++r)
      sc[(m0 + r) * SSTR + t * 16 + n] = acc[r];
  }
  __syncthreads();

  // ---- masked softmax: each lane owns one quarter of one of the 16 rows ----
  int   row  = tid & 15;
  int   quad = tid >> 4;               // 0..3
  int   c0   = quad * (SS / 4);
  const int* mrow = mask + ((size_t)b * SS + q0 + row) * SS;
  float* srow = sc + row * SSTR;
  const float invT   = 1.0f / 32.0f;   // 1/sqrt(D)
  const float NEGINF = -__builtin_inff();

  float mx = NEGINF;
#pragma unroll 4
  for (int i = 0; i < SS / 4; ++i) {
    int c = c0 + i;
    float s = mrow[c] ? NEGINF : srow[c] * invT;
    srow[c] = s;
    mx = fmaxf(mx, s);
  }
  mx = fmaxf(mx, __shfl_xor(mx, 16, 32));  // combine quarter pairs in-wave
  if (lane < 16) redmx[wave][row] = mx;
  __syncthreads();
  mx = fmaxf(redmx[0][row], redmx[1][row]);

  float sum = 0.f;
#pragma unroll 4
  for (int i = 0; i < SS / 4; ++i) {
    int c = c0 + i;
    float e = __expf(srow[c] - mx);
    srow[c] = e;
    sum += e;
  }
  sum += __shfl_xor(sum, 16, 32);
  if (lane < 16) redsm[wave][row] = sum;
  __syncthreads();
  sum = redsm[0][row] + redsm[1][row];
  float inv = 1.0f / sum;

  float* arow = attns + (hb * SS + q0 + row) * SS;
  bf16*  prow = pr + row * PSTR;
#pragma unroll 4
  for (int i = 0; i < SS / 4; ++i) {
    int c = c0 + i;
    float p = srow[c] * inv;
    arow[c] = p;             // f32 attention probs -> output tuple element 2
    prow[c] = (bf16)p;       // bf16 copy for the P@V WMMA
  }
  __syncthreads();

  // ---- out tile (16x64) = P (16x1024) @ V; waves take 2 N-tiles each ----
  f32x8 zero = {};
  f32x8 oacc[2];
  oacc[0] = zero;
  oacc[1] = zero;
  const bf16* vb0 = vt + (size_t)((wave * 2 + 0) * 16) * SS;
  const bf16* vb1 = vt + (size_t)((wave * 2 + 1) * 16) * SS;
#pragma unroll 4
  for (int kk = 0; kk < SS; kk += 32) {
    bf16x16 pa = load_a_frag((const bf16*)pr, PSTR, kk);
    bf16x16 u0 = load_bt_frag(vb0, SS, kk);
    bf16x16 u1 = load_bt_frag(vb1, SS, kk);
    oacc[0] = wmma_bf16(pa, u0, oacc[0]);
    oacc[1] = wmma_bf16(pa, u1, oacc[1]);
  }
  // store to (B, S, H*DV) layout so the output projection A is row-dense
  bf16* obase = out_heads + ((size_t)b * SS + q0) * (HH * DKV) + h * DKV;
#pragma unroll
  for (int jj = 0; jj < 2; ++jj) {
    int j = wave * 2 + jj;
#pragma unroll
    for (int r = 0; r < 8; ++r)
      obase[(size_t)(m0 + r) * (HH * DKV) + j * 16 + n] = (bf16)oacc[jj][r];
  }
}

// ---------------------------------------------------------------------------
// 4) Output projection: proj = Xh (8192x1024) @ proj_w^T (1024x1024), f32 out
// ---------------------------------------------------------------------------
__global__ void __launch_bounds__(128)
out_proj_gemm(const bf16* __restrict__ Xh, const bf16* __restrict__ Wb,
              float* __restrict__ proj) {
  int wave = threadIdx.x >> 5;
  int r0   = (blockIdx.x * (blockDim.x >> 5) + wave) * 16;
  int c0   = blockIdx.y * 64;
  const bf16* Xt = Xh + (size_t)r0 * DD;
  const bf16* W0 = Wb + (size_t)c0 * DD;

  f32x8 zero = {};
  f32x8 acc[4];
#pragma unroll
  for (int j = 0; j < 4; ++j) acc[j] = zero;

#pragma unroll 2
  for (int kk = 0; kk < DD; kk += 32) {
    bf16x16 a = load_a_frag(Xt, DD, kk);
    bf16x16 b[4];
#pragma unroll
    for (int j = 0; j < 4; ++j)
      b[j] = load_bt_frag(W0 + (size_t)(j * 16) * DD, DD, kk);
#pragma unroll
    for (int j = 0; j < 4; ++j)
      acc[j] = wmma_bf16(a, b[j], acc[j]);
  }

  int lane = threadIdx.x & 31;
  int n = lane & 15, m0 = (lane >> 4) << 3;
#pragma unroll
  for (int j = 0; j < 4; ++j)
#pragma unroll
    for (int r = 0; r < 8; ++r)
      proj[(size_t)(r0 + m0 + r) * DD + c0 + j * 16 + n] = acc[j][r];
}

// ---------------------------------------------------------------------------
// 5) bias + residual + LayerNorm, one row (1024) per 256-thread block
// ---------------------------------------------------------------------------
__global__ void __launch_bounds__(256)
ln_kernel(const float* __restrict__ proj, const float* __restrict__ pb,
          const float* __restrict__ resid, const float* __restrict__ g,
          const float* __restrict__ beta, float* __restrict__ out) {
  int row = blockIdx.x;
  int tid = threadIdx.x;
  __shared__ float red[256];

  float v[4];
  float s = 0.f, s2 = 0.f;
#pragma unroll
  for (int i = 0; i < 4; ++i) {
    int j = tid + i * 256;
    float x = proj[(size_t)row * DD + j] + pb[j] + resid[(size_t)row * DD + j];
    v[i] = x;
    s += x;
    s2 += x * x;
  }
  red[tid] = s;
  __syncthreads();
  for (int off = 128; off > 0; off >>= 1) {
    if (tid < off) red[tid] += red[tid + off];
    __syncthreads();
  }
  float mean = red[0] / (float)DD;
  __syncthreads();
  red[tid] = s2;
  __syncthreads();
  for (int off = 128; off > 0; off >>= 1) {
    if (tid < off) red[tid] += red[tid + off];
    __syncthreads();
  }
  float var = red[0] / (float)DD - mean * mean;
  float rs  = rsqrtf(var + 1e-5f);
#pragma unroll
  for (int i = 0; i < 4; ++i) {
    int j = tid + i * 256;
    out[(size_t)row * DD + j] = (v[i] - mean) * rs * g[j] + beta[j];
  }
}

// ---------------------------------------------------------------------------
// Host-side orchestration
// ---------------------------------------------------------------------------
extern "C" void kernel_launch(void* const* d_in, const int* in_sizes, int n_in,
                              void* d_out, int out_size, void* d_ws, size_t ws_size,
                              hipStream_t stream) {
  const float* q      = (const float*)d_in[0];
  const float* k      = (const float*)d_in[1];
  const float* v      = (const float*)d_in[2];
  const int*   mask   = (const int*)d_in[3];
  const float* w_qs   = (const float*)d_in[4];
  const float* w_ks   = (const float*)d_in[5];
  const float* w_vs   = (const float*)d_in[6];
  const float* proj_w = (const float*)d_in[7];
  const float* proj_b = (const float*)d_in[8];
  const float* ln_g   = (const float*)d_in[9];
  const float* ln_b   = (const float*)d_in[10];

  float* out_final = (float*)d_out;                              // (B,S,D)
  float* attns     = (float*)d_out + (size_t)BB * SS * DD;       // (H*B,S,S)

  // workspace carve-out
  size_t off = 0;
  auto alloc = [&](size_t bytes) -> void* {
    void* p = (char*)d_ws + off;
    off += (bytes + 255) & ~(size_t)255;
    return p;
  };
  const size_t nAct = (size_t)BB * SS * DD;        // 8M
  const size_t nW   = (size_t)HH * DD * DKV;       // 1M
  const size_t nHS  = (size_t)HH * BB * SS * DKV;  // 8M

  bf16* qbf  = (bf16*)alloc(nAct * 2);
  bf16* kbf  = (bf16*)alloc(nAct * 2);
  bf16* vbf  = (bf16*)alloc(nAct * 2);
  bf16* wqt  = (bf16*)alloc(nW * 2);
  bf16* wkt  = (bf16*)alloc(nW * 2);
  bf16* wvt  = (bf16*)alloc(nW * 2);
  bf16* pwbf = (bf16*)alloc((size_t)DD * HH * DKV * 2);
  bf16* q_s  = (bf16*)alloc(nHS * 2);              // [h][b][S][64]
  bf16* k_s  = (bf16*)alloc(nHS * 2);              // [h][b][S][64]
  bf16* v_tb = (bf16*)alloc(nHS * 2);              // [h][b][64][S]
  bf16* oh   = (bf16*)alloc(nAct * 2);             // (B,S,H*DV)
  float* prj = (float*)alloc(nAct * 4);            // (B*S, D) f32

  // 1) converts
  cvt_f32_bf16<<<2048, 256, 0, stream>>>(q, qbf, (int)nAct);
  cvt_f32_bf16<<<2048, 256, 0, stream>>>(k, kbf, (int)nAct);
  cvt_f32_bf16<<<2048, 256, 0, stream>>>(v, vbf, (int)nAct);
  cvt_w_transpose<<<1024, 256, 0, stream>>>(w_qs, wqt);
  cvt_w_transpose<<<1024, 256, 0, stream>>>(w_ks, wkt);
  cvt_w_transpose<<<1024, 256, 0, stream>>>(w_vs, wvt);
  cvt_f32_bf16<<<1024, 256, 0, stream>>>(proj_w, pwbf, (int)((size_t)DD * HH * DKV));

  // 2) per-head projections (grid: 512 M-tiles / 4 waves-per-block, H)
  dim3 pg(128, HH);
  head_proj_gemm<<<pg, 128, 0, stream>>>(qbf, wqt, q_s, 0);
  head_proj_gemm<<<pg, 128, 0, stream>>>(kbf, wkt, k_s, 0);
  head_proj_gemm<<<pg, 128, 0, stream>>>(vbf, wvt, v_tb, 1);

  // 3) attention (q-tiles, batch, heads); 2 waves per q-tile
  dim3 ag(SS / 16, BB, HH);
  attn_kernel<<<ag, 64, 0, stream>>>(q_s, k_s, v_tb, mask, attns, oh);

  // 4) output projection
  dim3 og(128, DD / 64);
  out_proj_gemm<<<og, 128, 0, stream>>>(oh, pwbf, prj);

  // 5) bias + residual + layernorm
  ln_kernel<<<BB * SS, 256, 0, stream>>>(prj, proj_b, q, ln_g, ln_b, out_final);

  (void)in_sizes; (void)n_in; (void)out_size; (void)ws_size;
}
